// CrossAttention_23751169147558
// MI455X (gfx1250) — matrix operand order repaired and checked by reading, not compiled, not run on hardware
//
#include <hip/hip_runtime.h>

#define DI __device__ __forceinline__

typedef _Float16 v16h __attribute__((ext_vector_type(16)));
typedef _Float16 v8h  __attribute__((ext_vector_type(8)));
typedef _Float16 v4h  __attribute__((ext_vector_type(4)));
typedef float    v8f  __attribute__((ext_vector_type(8)));

namespace {

constexpr int NB = 32;    // batch
constexpr int NT = 64;    // time
constexpr int NF = 128;   // feature axis
constexpr int ND = 64;    // model dim
constexpr int NH = 4;     // heads
constexpr int HK = 128;   // heads * key_dim
constexpr float SCALE = 0.17677669529663687f; // 1/sqrt(32)

// ---- WMMA fragment helpers (CDNA5 16x16x32 f16, wave32) -------------------

// A-matrix fragment: lane L(<16) holds row L, K = {kc+0..7, kc+16..23};
// lane L+16 holds row L, K = {kc+8..15, kc+24..31}.  m0 = &mat[row0*ld + kc].
DI v16h fragA(const _Float16* m0, int ld) {
  const int lane = threadIdx.x & 31;
  const _Float16* p = m0 + (lane & 15) * ld + ((lane >> 4) << 3);
  union { v16h v; v8h h[2]; } u;
  u.h[0] = *(const v8h*)(p);
  u.h[1] = *(const v8h*)(p + 16);
  return u.v;
}

// B-matrix fragment: lane n(<16) holds column n, K = {kc+0..7, kc+16..23}; lane n+16
// holds column n, K = {kc+8..15, kc+24..31}.  B kept transposed in LDS (Bt[n][k]),
// m0 = &Bt[col0*ld + kc]; two 8-half chunks per lane.
DI v16h fragB(const _Float16* m0, int ld) {
  const int lane = threadIdx.x & 31;
  const _Float16* p = m0 + (lane & 15) * ld + ((lane >> 4) << 3);
  union { v16h v; v8h h[2]; } u;
  u.h[0] = *(const v8h*)(p);
  u.h[1] = *(const v8h*)(p + 16);
  return u.v;
}

DI v8f wmma16(v16h a, v16h b, v8f c) {
  return __builtin_amdgcn_wmma_f32_16x16x32_f16(
      /*neg_a=*/false, a, /*neg_b=*/false, b,
      /*c_mod=*/(short)0, c, /*reuse_a=*/false, /*reuse_b=*/false);
}

// C/D layout: lane n(<16) col n rows M=r..; lane n+16 col n rows M=8+r.
template <bool WITH_BIAS>
DI void st_acc_f16(_Float16* mat, int ld, int rb, int cb, v8f c, const float* bias) {
  const int lane = threadIdx.x & 31;
  const int col = cb + (lane & 15);
  const int row = rb + ((lane & 16) ? 8 : 0);
  const float bv = WITH_BIAS ? bias[col] : 0.0f;
  union { v8f v; float f[8]; } u; u.v = c;
#pragma unroll
  for (int r = 0; r < 8; ++r) mat[(row + r) * ld + col] = (_Float16)(u.f[r] + bv);
}

// Transposed store (mat[col][row]); the 8 halves per lane are contiguous.
DI void st_acc_f16_T(_Float16* mat, int ld, int rb, int cb, v8f c, const float* bias) {
  const int lane = threadIdx.x & 31;
  const int col = cb + (lane & 15);
  const int row = rb + ((lane & 16) ? 8 : 0);
  const float bv = bias[col];
  union { v8f v; float f[8]; } uf; uf.v = c;
  union { v8h v; _Float16 h[8]; } uh;
#pragma unroll
  for (int r = 0; r < 8; ++r) uh.h[r] = (_Float16)(uf.f[r] + bv);
  *(v8h*)(mat + col * ld + row) = uh.v;   // ds_store_b128
}

DI void st_acc_f32(float* mat, int ld, int rb, int cb, v8f c) {
  const int lane = threadIdx.x & 31;
  const int col = cb + (lane & 15);
  const int row = rb + ((lane & 16) ? 8 : 0);
  union { v8f v; float f[8]; } u; u.v = c;
#pragma unroll
  for (int r = 0; r < 8; ++r) mat[(row + r) * ld + col] = u.f[r];
}

constexpr size_t smem_bytes(int R) {
  // wtq/wtk/wtv/wot (f16) + biases (f32) + Q/K/Vt (f16) + S (f32, xt overlaid)
  // + P (f16) + O (f16)
  return (size_t)67328 + (size_t)1024 * R + (size_t)6 * R * R;
}

} // namespace

// R = sequence length of the attended axis (64 temporal / 128 feature).
// TEMPORAL: write out = result + (to_b + fo_b); else out += result.
template <int R, bool TEMPORAL>
__global__ __launch_bounds__(256) void axial_mha_fused(
    const float* __restrict__ x,
    const float* __restrict__ wq, const float* __restrict__ bq,
    const float* __restrict__ wk, const float* __restrict__ bk,
    const float* __restrict__ wv, const float* __restrict__ bv,
    const float* __restrict__ wo, const float* __restrict__ bo,
    const float* __restrict__ bo2,
    float* __restrict__ out) {
  extern __shared__ char smem[];
  _Float16* wtq = (_Float16*)smem;            // [128][64]  w^T, scale folded for Q
  _Float16* wtk = wtq + HK * ND;              // [128][64]
  _Float16* wtv = wtk + HK * ND;              // [128][64]
  _Float16* wot = wtv + HK * ND;              // [64][128]  wo^T
  float* bsq = (float*)(wot + ND * HK);       // [128] (scaled)
  float* bsk = bsq + HK;                      // [128]
  float* bsv = bsk + HK;                      // [128]
  float* bso = bsv + HK;                      // [64]
  _Float16* Q  = (_Float16*)(bso + ND);       // [R][128]
  _Float16* Kl = Q + R * HK;                  // [R][128]
  _Float16* Vt = Kl + R * HK;                 // [128][R]  (V transposed)
  float* S = (float*)(Vt + HK * R);           // [R][R] scores (f32)
  _Float16* xt = (_Float16*)S;                // [R][64] x tile (overlaid, dead after QKV)
  _Float16* P  = (_Float16*)(S + R * R);      // [R][R] probs (f16)
  _Float16* O  = P + R * R;                   // [R][128] concat head outputs

  const int tid = threadIdx.x;
  const int wid = tid >> 5;                   // 8 waves
  const int lane = tid & 31;

  // --- block -> slice decode -------------------------------------------------
  int baseoff, rstride;
  if (TEMPORAL) {              // block = (b, f); rows are t
    const int b = blockIdx.x / NF, f = blockIdx.x % NF;
    baseoff = (b * NT * NF + f) * ND;
    rstride = NF * ND;
  } else {                     // block = (b, t); rows are f (contiguous tile)
    baseoff = blockIdx.x * NF * ND;
    rstride = ND;
  }

  // --- stage weights (transposed, f16) + biases + x tile ---------------------
  // wq/wk/wv: [64][128] row-major -> wt*: [128][64]; vectorized global reads.
  for (int i4 = tid; i4 < (HK * ND) / 4; i4 += 256) {
    const int d = i4 >> 5, n4 = (i4 & 31) * 4;
    const float4 q4 = *(const float4*)(wq + i4 * 4);
    const float4 k4 = *(const float4*)(wk + i4 * 4);
    const float4 v4 = *(const float4*)(wv + i4 * 4);
#pragma unroll
    for (int j = 0; j < 4; ++j) {
      wtq[(n4 + j) * ND + d] = (_Float16)(((const float*)&q4)[j] * SCALE);
      wtk[(n4 + j) * ND + d] = (_Float16)(((const float*)&k4)[j]);
      wtv[(n4 + j) * ND + d] = (_Float16)(((const float*)&v4)[j]);
    }
  }
  // wo: [128][64] -> wot: [64][128]
  for (int i4 = tid; i4 < (HK * ND) / 4; i4 += 256) {
    const int hk = i4 >> 4, d4 = (i4 & 15) * 4;
    const float4 o4 = *(const float4*)(wo + i4 * 4);
#pragma unroll
    for (int j = 0; j < 4; ++j)
      wot[(d4 + j) * HK + hk] = (_Float16)(((const float*)&o4)[j]);
  }
  for (int i = tid; i < HK; i += 256) {
    bsq[i] = bq[i] * SCALE; bsk[i] = bk[i]; bsv[i] = bv[i];
  }
  if (tid < ND) bso[tid] = TEMPORAL ? (bo[tid] + bo2[tid]) : 0.0f;
  // x tile -> f16 (16B-aligned float4 loads, 8B LDS stores)
  for (int i4 = tid; i4 < (R * ND) / 4; i4 += 256) {
    const int r = i4 >> 4, c4 = (i4 & 15) * 4;
    const float4 xv = *(const float4*)(x + baseoff + r * rstride + c4);
    v4h hv;
    hv.x = (_Float16)xv.x; hv.y = (_Float16)xv.y;
    hv.z = (_Float16)xv.z; hv.w = (_Float16)xv.w;
    *(v4h*)(xt + i4 * 4) = hv;
  }
  __syncthreads();

  constexpr int MT = R / 16;

  // --- QKV projection: [R,64] @ [64,128], fused for q/k/v --------------------
  for (int t = wid; t < MT * 8; t += 8) {
    const int mt = t >> 3, nt = t & 7;
    v8f cq = {0,0,0,0,0,0,0,0}, ck = cq, cv = cq;
#pragma unroll
    for (int kc = 0; kc < ND; kc += 32) {
      const v16h a = fragA(xt + mt * 16 * ND + kc, ND);
      cq = wmma16(a, fragB(wtq + nt * 16 * ND + kc, ND), cq);
      ck = wmma16(a, fragB(wtk + nt * 16 * ND + kc, ND), ck);
      cv = wmma16(a, fragB(wtv + nt * 16 * ND + kc, ND), cv);
    }
    st_acc_f16<true>(Q,  HK, mt * 16, nt * 16, cq, bsq);
    st_acc_f16<true>(Kl, HK, mt * 16, nt * 16, ck, bsk);
    st_acc_f16_T(Vt, R, mt * 16, nt * 16, cv, bsv);   // Vt[n][m]
  }
  __syncthreads();  // xt dead; S region free

  // --- per-head attention ----------------------------------------------------
  for (int h = 0; h < NH; ++h) {
    // scores S = Q_h @ K_h^T  (head dim 32 -> single WMMA step)
    for (int t = wid; t < MT * MT; t += 8) {
      const int mt = t / MT, nt = t % MT;
      v8f c = {0,0,0,0,0,0,0,0};
      c = wmma16(fragA(Q + mt * 16 * HK + h * 32, HK),
                 fragB(Kl + nt * 16 * HK + h * 32, HK), c);
      st_acc_f32(S, R, mt * 16, nt * 16, c);
    }
    __syncthreads();   // also fences previous head's PV reads of P

    // row softmax -> P (f16); single exp per element
    for (int r = tid; r < R; r += 256) {
      float mx = -1e30f;
      for (int c = 0; c < R; ++c) mx = fmaxf(mx, S[r * R + c]);
      float sum = 0.0f;
      for (int c = 0; c < R; ++c) {
        const float e = __expf(S[r * R + c] - mx);
        sum += e;
        P[r * R + c] = (_Float16)e;
      }
      const float rs = 1.0f / sum;
      for (int c = 0; c < R; ++c)
        P[r * R + c] = (_Float16)((float)P[r * R + c] * rs);
    }
    __syncthreads();

    // O[:, h*32 : h*32+32] = P @ V_h   (B operand = Vt rows, contiguous)
    for (int t = wid; t < MT * 2; t += 8) {
      const int mt = t >> 1, nt = t & 1;
      v8f c = {0,0,0,0,0,0,0,0};
#pragma unroll
      for (int kc = 0; kc < R; kc += 32) {
        c = wmma16(fragA(P + mt * 16 * R + kc, R),
                   fragB(Vt + (h * 32 + nt * 16) * R + kc, R), c);
      }
      st_acc_f16<false>(O, HK, mt * 16, h * 32 + nt * 16, c, nullptr);
    }
  }
  __syncthreads();

  // --- output projection: [R,128] @ [128,64] -> global -----------------------
  for (int t = wid; t < MT * 4; t += 8) {
    const int mt = t >> 2, nt = t & 3;
    v8f c = {0,0,0,0,0,0,0,0};
#pragma unroll
    for (int kc = 0; kc < HK; kc += 32) {
      c = wmma16(fragA(O + mt * 16 * HK + kc, HK),
                 fragB(wot + nt * 16 * HK + kc, HK), c);
    }
    const int col = nt * 16 + (lane & 15);
    const int row = mt * 16 + ((lane & 16) ? 8 : 0);
    union { v8f v; float f[8]; } u; u.v = c;
    if (TEMPORAL) {
      const float bv = bso[col];
#pragma unroll
      for (int r = 0; r < 8; ++r)
        out[baseoff + (row + r) * rstride + col] = u.f[r] + bv;
    } else {
#pragma unroll
      for (int r = 0; r < 8; ++r) {
        const int idx = baseoff + (row + r) * rstride + col;
        out[idx] += u.f[r];   // disjoint per block; stream-ordered after temporal
      }
    }
  }
}

extern "C" void kernel_launch(void* const* d_in, const int* in_sizes, int n_in,
                              void* d_out, int out_size, void* d_ws, size_t ws_size,
                              hipStream_t stream) {
  (void)in_sizes; (void)n_in; (void)out_size; (void)d_ws; (void)ws_size;
  const float* x    = (const float*)d_in[0];
  const float* tq_w = (const float*)d_in[1];
  const float* tq_b = (const float*)d_in[2];
  const float* tk_w = (const float*)d_in[3];
  const float* tk_b = (const float*)d_in[4];
  const float* tv_w = (const float*)d_in[5];
  const float* tv_b = (const float*)d_in[6];
  const float* fq_w = (const float*)d_in[7];
  const float* fq_b = (const float*)d_in[8];
  const float* fk_w = (const float*)d_in[9];
  const float* fk_b = (const float*)d_in[10];
  const float* fv_w = (const float*)d_in[11];
  const float* fv_b = (const float*)d_in[12];
  const float* to_w = (const float*)d_in[13];
  const float* to_b = (const float*)d_in[14];
  const float* fo_w = (const float*)d_in[15];
  const float* fo_b = (const float*)d_in[16];
  float* out = (float*)d_out;

  const size_t smT = smem_bytes(64);    // ~154 KB -> 2 workgroups / WGP
  const size_t smF = smem_bytes(128);   // ~290 KB -> needs CDNA5's 320 KB LDS/WGP
  (void)hipFuncSetAttribute(reinterpret_cast<const void*>(&axial_mha_fused<64, true>),
                            hipFuncAttributeMaxDynamicSharedMemorySize, (int)smT);
  (void)hipFuncSetAttribute(reinterpret_cast<const void*>(&axial_mha_fused<128, false>),
                            hipFuncAttributeMaxDynamicSharedMemorySize, (int)smF);

  // Temporal branch: writes out = temporal(x) + to_b + fo_b
  axial_mha_fused<64, true><<<NB * NF, 256, smT, stream>>>(
      x, tq_w, tq_b, tk_w, tk_b, tv_w, tv_b, to_w, to_b, fo_b, out);
  // Feature branch: accumulates out += feature(x)
  axial_mha_fused<128, false><<<NB * NT, 256, smF, stream>>>(
      x, fq_w, fq_b, fk_w, fk_b, fv_w, fv_b, fo_w, fo_b, to_b, out);
}
